// TransMIL_21990232555652
// MI455X (gfx1250) — matrix operand order repaired and checked by reading, not compiled, not run on hardware
//
#include <hip/hip_runtime.h>

// ---------------------------------------------------------------------------
// CDNA5 (gfx1250) implementation: wave32, 32x32 tile/wave, 4x v_wmma_f32_16x16x32_bf16
// ---------------------------------------------------------------------------

typedef __attribute__((ext_vector_type(16))) __bf16 v16bf;
typedef __attribute__((ext_vector_type(8)))  float  v8f;

constexpr int  B_    = 2;
constexpr int  C_    = 512;
constexpr int  N0_   = 8192;
constexpr int  HS_   = 91;
constexpr int  NSQ_  = 8281;   // 91*91
constexpr int  N2_   = 8282;   // +cls
constexpr int  NPAD_ = 8448;   // attention padded length
constexpr int  PADF_ = 166;    // front pad rows
constexpr int  LM_   = 33;     // rows per landmark chunk (npad/M)
constexpr int  NH_   = 8;
constexpr int  DH_   = 64;
constexpr int  MM_   = 256;    // landmark count
constexpr int  K3_   = 1536;   // 3*C

// workspace layout (float offsets)
constexpr long long OFF_STREAM = 0;
constexpr long long OFF_PAD    = OFF_STREAM + (long long)B_ * N2_ * C_;    // also head-out buffer
constexpr long long OFF_QKV    = OFF_PAD    + (long long)B_ * NPAD_ * C_;
constexpr long long OFF_QL     = OFF_QKV    + (long long)B_ * NPAD_ * K3_;
constexpr long long OFF_KL     = OFF_QL  + 16LL * MM_ * DH_;
constexpr long long OFF_A2     = OFF_KL  + 16LL * MM_ * DH_;
constexpr long long OFF_ZA     = OFF_A2  + 16LL * MM_ * MM_;
constexpr long long OFF_ZB     = OFF_ZA  + 16LL * MM_ * MM_;
constexpr long long OFF_T1     = OFF_ZB  + 16LL * MM_ * MM_;
constexpr long long OFF_M2     = OFF_T1  + 16LL * MM_ * MM_;
constexpr long long OFF_M3     = OFF_M2  + 16LL * MM_ * MM_;
constexpr long long OFF_A3V    = OFF_M3  + 16LL * MM_ * MM_;
constexpr long long OFF_ZAV    = OFF_A3V + 16LL * MM_ * DH_;
constexpr long long OFF_SCAL   = OFF_ZAV + 16LL * MM_ * DH_;
constexpr long long OFF_BIG    = OFF_SCAL + 64;                            // scores / conv scratch
constexpr long long OFF_BF     = OFF_BIG + 16LL * NPAD_ * MM_;             // bf16 weight area (as floats)

// bf16-element offsets inside the bf16 area
constexpr long long BW_FC1  = 0;                              // 512*2048
constexpr long long BW_QKV1 = BW_FC1  + 1048576;              // 1536*512
constexpr long long BW_QKV2 = BW_QKV1 + 786432;
constexpr long long BW_OUT1 = BW_QKV2 + 786432;               // 512*512
constexpr long long BW_OUT2 = BW_OUT1 + 262144;
constexpr long long BW_KL   = BW_OUT2 + 262144;               // 16*256*64

static __device__ __forceinline__ __bf16 tobf(float x)  { return (__bf16)x; }
static __device__ __forceinline__ __bf16 tobf(__bf16 x) { return x; }

#define WMMA_BF16(A, Bv, Cacc) \
  __builtin_amdgcn_wmma_f32_16x16x32_bf16(false, (A), false, (Bv), (short)0, (Cacc), false, false)

// ---------------------------------------------------------------------------
// GEMM 1: C = act(oscale * A @ W^T + bias)
// A: MxK row-major f32; W: NxK row-major (f32 or bf16, templated).
// One wave32 computes a 32x32 tile: 2 A-frags x 2 B-frags -> 4 WMMA per k-step.
// Batched over grid.y: z -> (b = z/nheads, h = z%nheads).
// Output rows < rowskip dropped; remaining rows land at (m - rowskip); accum adds.
// ---------------------------------------------------------------------------
template <typename WT>
__global__ __launch_bounds__(256) void gemm_awt_kernel(
    const float* __restrict__ Abase, long long sAb, long long sAh, int lda,
    const WT* __restrict__ Wbase, long long sWb, long long sWh, int ldw,
    const float* __restrict__ bias,
    float* __restrict__ Cbase, long long sCb, long long sCh, int ldc,
    int M, int N, int K, int nheads,
    float oscale, int relu, int rowskip, int accum)
{
  const int z = blockIdx.y;
  const int b = z / nheads, h = z % nheads;
  const float* A  = Abase + (long long)b * sAb + (long long)h * sAh;
  const WT*    W  = Wbase + (long long)b * sWb + (long long)h * sWh;
  float*       Cp = Cbase + (long long)b * sCb + (long long)h * sCh;

  const int mt = M >> 5, nt = N >> 5;
  const int tile = blockIdx.x * 8 + threadIdx.y;
  if (tile >= mt * nt) return;
  const int tm = tile / nt, tn = tile % nt;

  const int lane = threadIdx.x;
  const int r  = lane & 15;
  const int hf = lane >> 4;
  const float* a0 = A + (long long)(tm * 32 + r) * lda;
  const float* a1 = a0 + 16LL * lda;
  const WT* w0 = W + (long long)(tn * 32 + r) * ldw;
  const WT* w1 = w0 + 16LL * ldw;

  v8f c00 = {}, c01 = {}, c10 = {}, c11 = {};
  for (int k0 = 0; k0 < K; k0 += 32) {
    __builtin_prefetch((const void*)(a0 + k0 + 32), 0, 0);
    __builtin_prefetch((const void*)(a1 + k0 + 32), 0, 0);
    v16bf av0, av1, bv0, bv1;
#pragma unroll
    for (int i = 0; i < 8; ++i) {
      av0[i]     = tobf(a0[k0 + hf * 16 + i]);       // K = k0 + hf*16 + i
      av0[i + 8] = tobf(a0[k0 + hf * 16 + 8 + i]);   // K = k0 + hf*16 + 8 + i
      av1[i]     = tobf(a1[k0 + hf * 16 + i]);
      av1[i + 8] = tobf(a1[k0 + hf * 16 + 8 + i]);
    }
#pragma unroll
    for (int i = 0; i < 16; ++i) {
      bv0[i] = tobf(w0[k0 + hf * 16 + i]);           // B[k][n] = W[n][k], 16 contiguous K
      bv1[i] = tobf(w1[k0 + hf * 16 + i]);
    }
    c00 = WMMA_BF16(av0, bv0, c00);
    c01 = WMMA_BF16(av0, bv1, c01);
    c10 = WMMA_BF16(av1, bv0, c10);
    c11 = WMMA_BF16(av1, bv1, c11);
  }

  const int n0 = tn * 32 + r, n1 = n0 + 16;
  const float bb0 = bias ? bias[n0] : 0.f;
  const float bb1 = bias ? bias[n1] : 0.f;
  auto emit = [&](int m_g, int n_g, float accv, float bval) {
    if (m_g < rowskip) return;
    float v = accv * oscale + bval;
    if (relu) v = fmaxf(v, 0.f);
    float* cp = Cp + (long long)(m_g - rowskip) * ldc + n_g;
    if (accum) *cp += v; else *cp = v;
  };
#pragma unroll
  for (int j = 0; j < 8; ++j) {
    const int m0 = tm * 32 + j + 8 * hf;   // C layout: vgpr j, halves hold M and M+8
    const int m1 = m0 + 16;
    emit(m0, n0, c00[j], bb0);
    emit(m0, n1, c01[j], bb1);
    emit(m1, n0, c10[j], bb0);
    emit(m1, n1, c11[j], bb1);
  }
}

// ---------------------------------------------------------------------------
// GEMM 2: C = o_scale * (A @ (b_scale*B + b_diag*I)) + o_diag*I
// A: MxK row-major; B: KxN row-major (strided loads). 32x32 tile per wave.
// Covers all Newton-iteration steps, a3@v, z@a3v, a1@zav.
// ---------------------------------------------------------------------------
__global__ __launch_bounds__(256) void gemm_ab_kernel(
    const float* __restrict__ Abase, long long sAb, long long sAh, int lda,
    const float* __restrict__ Bbase, long long sBb, long long sBh, int ldb,
    float* __restrict__ Cbase, long long sCb, long long sCh, int ldc,
    int M, int N, int K, int nheads,
    float b_scale, float b_diag, float o_scale, float o_diag)
{
  const int z = blockIdx.y;
  const int b = z / nheads, h = z % nheads;
  const float* A  = Abase + (long long)b * sAb + (long long)h * sAh;
  const float* Bp = Bbase + (long long)b * sBb + (long long)h * sBh;
  float*       Cp = Cbase + (long long)b * sCb + (long long)h * sCh;

  const int mt = M >> 5, nt = N >> 5;
  const int tile = blockIdx.x * 8 + threadIdx.y;
  if (tile >= mt * nt) return;
  const int tm = tile / nt, tn = tile % nt;

  const int lane = threadIdx.x;
  const int r  = lane & 15;
  const int hf = lane >> 4;
  const float* a0 = A + (long long)(tm * 32 + r) * lda;
  const float* a1 = a0 + 16LL * lda;
  const int n0 = tn * 32 + r, n1 = n0 + 16;

  v8f c00 = {}, c01 = {}, c10 = {}, c11 = {};
  for (int k0 = 0; k0 < K; k0 += 32) {
    __builtin_prefetch((const void*)(a0 + k0 + 32), 0, 0);
    __builtin_prefetch((const void*)(a1 + k0 + 32), 0, 0);
    v16bf av0, av1, bv0, bv1;
#pragma unroll
    for (int i = 0; i < 8; ++i) {
      av0[i]     = tobf(a0[k0 + hf * 16 + i]);
      av0[i + 8] = tobf(a0[k0 + hf * 16 + 8 + i]);
      av1[i]     = tobf(a1[k0 + hf * 16 + i]);
      av1[i + 8] = tobf(a1[k0 + hf * 16 + 8 + i]);
    }
#pragma unroll
    for (int i = 0; i < 16; ++i) {
      const int kk = k0 + hf * 16 + i;
      const float* brow = Bp + (long long)kk * ldb;
      float x0 = b_scale * brow[n0];
      float x1 = b_scale * brow[n1];
      if (kk == n0) x0 += b_diag;
      if (kk == n1) x1 += b_diag;
      bv0[i] = tobf(x0);
      bv1[i] = tobf(x1);
    }
    c00 = WMMA_BF16(av0, bv0, c00);
    c01 = WMMA_BF16(av0, bv1, c01);
    c10 = WMMA_BF16(av1, bv0, c10);
    c11 = WMMA_BF16(av1, bv1, c11);
  }

#pragma unroll
  for (int j = 0; j < 8; ++j) {
    const int m0 = tm * 32 + j + 8 * hf;
    const int m1 = m0 + 16;
    float v00 = o_scale * c00[j]; if (m0 == n0) v00 += o_diag;
    float v01 = o_scale * c01[j]; if (m0 == n1) v01 += o_diag;
    float v10 = o_scale * c10[j]; if (m1 == n0) v10 += o_diag;
    float v11 = o_scale * c11[j]; if (m1 == n1) v11 += o_diag;
    Cp[(long long)m0 * ldc + n0] = v00;
    Cp[(long long)m0 * ldc + n1] = v01;
    Cp[(long long)m1 * ldc + n0] = v10;
    Cp[(long long)m1 * ldc + n1] = v11;
  }
}

// ---------------------------------------------------------------------------
// f32 -> bf16 weight conversion (once per call for stationary operands)
// ---------------------------------------------------------------------------
__global__ void f32_to_bf16_kernel(const float* __restrict__ src, __bf16* __restrict__ dst,
                                   long long n)
{
  const long long i = (long long)blockIdx.x * 256 + threadIdx.x;
  if (i < n) dst[i] = (__bf16)src[i];
}

// ---------------------------------------------------------------------------
// cls token + wraparound rows (stream[0]=cls; stream[8193+j]=stream[1+j], j<89)
// ---------------------------------------------------------------------------
__global__ void fixup_kernel(const float* __restrict__ cls, float* __restrict__ stream)
{
  const int idx = blockIdx.x * 256 + threadIdx.x;
  if (idx >= B_ * 90 * C_) return;
  const int c = idx % C_;
  const int r = (idx / C_) % 90;
  const int b = idx / (90 * C_);
  float* sb = stream + (long long)b * N2_ * C_;
  if (r == 0) sb[c] = cls[c];
  else        sb[(long long)(N0_ + r) * C_ + c] = sb[(long long)r * C_ + c];
}

// ---------------------------------------------------------------------------
// layernorm + front-pad: out rows [0,166) = 0, out row p (>=166) = LN(stream row p-166)
// ---------------------------------------------------------------------------
__global__ void ln_pad_kernel(const float* __restrict__ stream,
                              const float* __restrict__ w, const float* __restrict__ bb,
                              float* __restrict__ out)
{
  const int rowp = blockIdx.x % NPAD_;
  const int b    = blockIdx.x / NPAD_;
  float* o = out + ((long long)b * NPAD_ + rowp) * C_;
  const int tid = threadIdx.x;
  if (rowp < PADF_) {
    for (int c = tid; c < C_; c += 256) o[c] = 0.f;
    return;
  }
  const float* x = stream + ((long long)b * N2_ + (rowp - PADF_)) * C_;
  __shared__ float red[256];
  float s = 0.f;
  for (int c = tid; c < C_; c += 256) s += x[c];
  red[tid] = s; __syncthreads();
  for (int st = 128; st > 0; st >>= 1) { if (tid < st) red[tid] += red[tid + st]; __syncthreads(); }
  const float mu = red[0] / (float)C_;
  __syncthreads();
  float v = 0.f;
  for (int c = tid; c < C_; c += 256) { float d = x[c] - mu; v += d * d; }
  red[tid] = v; __syncthreads();
  for (int st = 128; st > 0; st >>= 1) { if (tid < st) red[tid] += red[tid + st]; __syncthreads(); }
  const float rstd = rsqrtf(red[0] / (float)C_ + 1e-5f);
  for (int c = tid; c < C_; c += 256) o[c] = (x[c] - mu) * rstd * w[c] + bb[c];
}

// ---------------------------------------------------------------------------
// landmarks: ql (scaled by 1/8) f32, kl f32 + bf16 copy (reused as GEMM W operand)
// ---------------------------------------------------------------------------
__global__ void landmarks_kernel(const float* __restrict__ qkv,
                                 float* __restrict__ ql, float* __restrict__ kl,
                                 __bf16* __restrict__ klbf)
{
  const long long idx = (long long)blockIdx.x * 256 + threadIdx.x;  // 16*256*64
  if (idx >= 16LL * MM_ * DH_) return;
  const int d  = (int)(idx & 63);
  const int m  = (int)((idx >> 6) & 255);
  const int zh = (int)(idx >> 14);
  const int b = zh >> 3, h = zh & 7;
  const float* base = qkv + ((long long)b * NPAD_ + (long long)m * LM_) * K3_ + h * DH_ + d;
  float sq = 0.f, sk = 0.f;
  for (int j = 0; j < LM_; ++j) {
    sq += base[(long long)j * K3_];
    sk += base[(long long)j * K3_ + C_];
  }
  const float qv = sq * (1.f / 33.f) * 0.125f;
  const float kv = sk * (1.f / 33.f);
  ql[idx] = qv;
  kl[idx] = kv;
  klbf[idx] = (__bf16)kv;
}

// ---------------------------------------------------------------------------
// generic in-place row softmax
// ---------------------------------------------------------------------------
__global__ void row_softmax_kernel(float* __restrict__ data, int ncols)
{
  const long long row = blockIdx.x;
  float* p = data + row * (long long)ncols;
  __shared__ float red[256];
  const int tid = threadIdx.x;
  float mx = -3.4e38f;
  for (int c = tid; c < ncols; c += 256) mx = fmaxf(mx, p[c]);
  red[tid] = mx; __syncthreads();
  for (int st = 128; st > 0; st >>= 1) { if (tid < st) red[tid] = fmaxf(red[tid], red[tid + st]); __syncthreads(); }
  mx = red[0]; __syncthreads();
  float s = 0.f;
  for (int c = tid; c < ncols; c += 256) { float e = __expf(p[c] - mx); p[c] = e; s += e; }
  red[tid] = s; __syncthreads();
  for (int st = 128; st > 0; st >>= 1) { if (tid < st) red[tid] += red[tid + st]; __syncthreads(); }
  const float inv = 1.f / red[0];
  __syncthreads();
  for (int c = tid; c < ncols; c += 256) p[c] *= inv;
}

// ---------------------------------------------------------------------------
// pinv init helpers: global max of |a2| row-sums and col-sums, then z0 = a2^T/(s0*s1)
// ---------------------------------------------------------------------------
__global__ void reset_scal_kernel(float* s) { if (threadIdx.x < 2 && blockIdx.x == 0) s[threadIdx.x] = 0.f; }

__global__ void absrowmax_kernel(const float* __restrict__ a2, float* __restrict__ s)
{
  const int idx = blockIdx.x * 256 + threadIdx.x;
  if (idx >= 16 * MM_) return;
  const float* p = a2 + (long long)idx * MM_;
  float sum = 0.f;
  for (int i = 0; i < MM_; ++i) sum += fabsf(p[i]);
  atomicMax((unsigned int*)&s[0], __float_as_uint(sum));
}

__global__ void abscolmax_kernel(const float* __restrict__ a2, float* __restrict__ s)
{
  const int idx = blockIdx.x * 256 + threadIdx.x;
  if (idx >= 16 * MM_) return;
  const int zh = idx >> 8, col = idx & 255;
  const float* p = a2 + (long long)zh * MM_ * MM_ + col;
  float sum = 0.f;
  for (int i = 0; i < MM_; ++i) sum += fabsf(p[(long long)i * MM_]);
  atomicMax((unsigned int*)&s[1], __float_as_uint(sum));
}

__global__ void init_z_kernel(const float* __restrict__ a2, const float* __restrict__ s,
                              float* __restrict__ z)
{
  const long long idx = (long long)blockIdx.x * 256 + threadIdx.x;  // 16*256*256
  if (idx >= 16LL * MM_ * MM_) return;
  const int zh = (int)(idx >> 16);
  const int i  = (int)((idx >> 8) & 255);
  const int j  = (int)(idx & 255);
  const float scale = 1.f / (s[0] * s[1]);
  z[idx] = a2[(long long)zh * MM_ * MM_ + (long long)j * MM_ + i] * scale;
}

// ---------------------------------------------------------------------------
// residual depthwise conv along sequence (kernel 33, pad 16) added to head-out
// ---------------------------------------------------------------------------
__global__ void resconv_kernel(const float* __restrict__ qkv, const float* __restrict__ rw,
                               float* __restrict__ headout)
{
  const long long idx = (long long)blockIdx.x * 256 + threadIdx.x;  // B*NPAD*C
  if (idx >= (long long)B_ * NPAD_ * C_) return;
  const int c = (int)(idx & 511);
  const long long t = (idx >> 9) % NPAD_;
  const int b = (int)(idx / ((long long)NPAD_ * C_));
  const int h = c >> 6;
  const float* v = qkv + (long long)b * NPAD_ * K3_ + 2 * C_ + c;
  float acc = 0.f;
  for (int j = 0; j < 33; ++j) {
    const long long tt = t + j - 16;
    if (tt >= 0 && tt < NPAD_) acc += rw[h * 33 + j] * v[tt * K3_];
  }
  headout[idx] += acc;
}

// ---------------------------------------------------------------------------
// depthwise 7x7 + identity + 5x5 + 3x3 mixer over 91x91 grid (reads stream rows 1..)
// ---------------------------------------------------------------------------
__global__ void dwconv_kernel(const float* __restrict__ stream,
                              const float* __restrict__ w7, const float* __restrict__ b7,
                              const float* __restrict__ w5, const float* __restrict__ b5,
                              const float* __restrict__ w3, const float* __restrict__ b3,
                              float* __restrict__ outbuf)
{
  const long long idx = (long long)blockIdx.x * 256 + threadIdx.x;  // B*NSQ*C
  if (idx >= (long long)B_ * NSQ_ * C_) return;
  const int c  = (int)(idx & 511);
  const int hw = (int)((idx >> 9) % NSQ_);
  const int b  = (int)(idx / ((long long)NSQ_ * C_));
  const int y = hw / HS_, x = hw % HS_;
  const float* sbase = stream + ((long long)b * N2_ + 1) * C_ + c;  // feat row n at stride C

  float acc = sbase[(long long)hw * C_];          // + cnn (identity)
  acc += b7[c] + b5[c] + b3[c];
  for (int dy = -3; dy <= 3; ++dy) {
    const int yy = y + dy; if (yy < 0 || yy >= HS_) continue;
    for (int dx = -3; dx <= 3; ++dx) {
      const int xx = x + dx; if (xx < 0 || xx >= HS_) continue;
      acc += w7[c * 49 + (dy + 3) * 7 + (dx + 3)] * sbase[(long long)(yy * HS_ + xx) * C_];
    }
  }
  for (int dy = -2; dy <= 2; ++dy) {
    const int yy = y + dy; if (yy < 0 || yy >= HS_) continue;
    for (int dx = -2; dx <= 2; ++dx) {
      const int xx = x + dx; if (xx < 0 || xx >= HS_) continue;
      acc += w5[c * 25 + (dy + 2) * 5 + (dx + 2)] * sbase[(long long)(yy * HS_ + xx) * C_];
    }
  }
  for (int dy = -1; dy <= 1; ++dy) {
    const int yy = y + dy; if (yy < 0 || yy >= HS_) continue;
    for (int dx = -1; dx <= 1; ++dx) {
      const int xx = x + dx; if (xx < 0 || xx >= HS_) continue;
      acc += w3[c * 9 + (dy + 1) * 3 + (dx + 1)] * sbase[(long long)(yy * HS_ + xx) * C_];
    }
  }
  outbuf[idx] = acc;
}

__global__ void copyback_kernel(const float* __restrict__ buf, float* __restrict__ stream)
{
  const long long idx = (long long)blockIdx.x * 256 + threadIdx.x;  // B*NSQ*C
  if (idx >= (long long)B_ * NSQ_ * C_) return;
  const int c  = (int)(idx & 511);
  const long long n = (idx >> 9) % NSQ_;
  const int b  = (int)(idx / ((long long)NSQ_ * C_));
  stream[((long long)b * N2_ + 1 + n) * C_ + c] = buf[idx];
}

// ---------------------------------------------------------------------------
// final: pooled = LN(stream row 0); out = pooled @ fc2_w^T + fc2_b
// ---------------------------------------------------------------------------
__global__ void final_kernel(const float* __restrict__ stream,
                             const float* __restrict__ lnw, const float* __restrict__ lnb,
                             const float* __restrict__ fw, const float* __restrict__ fb,
                             float* __restrict__ out)
{
  const int b = blockIdx.x;
  const float* x = stream + (long long)b * N2_ * C_;  // row 0
  __shared__ float red[256];
  __shared__ float pooled[C_];
  const int tid = threadIdx.x;
  float s = 0.f;
  for (int c = tid; c < C_; c += 256) s += x[c];
  red[tid] = s; __syncthreads();
  for (int st = 128; st > 0; st >>= 1) { if (tid < st) red[tid] += red[tid + st]; __syncthreads(); }
  const float mu = red[0] / (float)C_;
  __syncthreads();
  float v = 0.f;
  for (int c = tid; c < C_; c += 256) { float d = x[c] - mu; v += d * d; }
  red[tid] = v; __syncthreads();
  for (int st = 128; st > 0; st >>= 1) { if (tid < st) red[tid] += red[tid + st]; __syncthreads(); }
  const float rstd = rsqrtf(red[0] / (float)C_ + 1e-5f);
  __syncthreads();
  for (int c = tid; c < C_; c += 256) pooled[c] = (x[c] - mu) * rstd * lnw[c] + lnb[c];
  __syncthreads();
  for (int o = 0; o < 2; ++o) {
    float p = 0.f;
    for (int c = tid; c < C_; c += 256) p += pooled[c] * fw[o * C_ + c];
    red[tid] = p; __syncthreads();
    for (int st = 128; st > 0; st >>= 1) { if (tid < st) red[tid] += red[tid + st]; __syncthreads(); }
    if (tid == 0) out[b * 2 + o] = red[0] + fb[o];
    __syncthreads();
  }
}

// ---------------------------------------------------------------------------
// host-side launch helpers
// ---------------------------------------------------------------------------
static inline dim3 gemm_grid(int M, int N, int batches) {
  const int tiles = (M / 32) * (N / 32);
  return dim3((unsigned)((tiles + 7) / 8), (unsigned)batches, 1);
}
static const dim3 GEMM_BLOCK(32, 8, 1);
static inline int eb(long long n) { return (int)((n + 255) / 256); }  // elementwise blocks

static void attention_block(float* ws,
                            const float* lnw, const float* lnb,
                            const __bf16* qkvw_bf, const __bf16* outw_bf,
                            const float* outb, const float* resw,
                            hipStream_t st)
{
  float* stream = ws + OFF_STREAM;
  float* pad    = ws + OFF_PAD;     // LN output -> later head-out buffer
  float* qkv    = ws + OFF_QKV;
  float* ql     = ws + OFF_QL;
  float* kl     = ws + OFF_KL;
  float* a2     = ws + OFF_A2;
  float* zA     = ws + OFF_ZA;
  float* zB     = ws + OFF_ZB;
  float* t1     = ws + OFF_T1;
  float* m2     = ws + OFF_M2;
  float* m3     = ws + OFF_M3;
  float* a3v    = ws + OFF_A3V;
  float* zav    = ws + OFF_ZAV;
  float* scal   = ws + OFF_SCAL;
  float* big    = ws + OFF_BIG;
  __bf16* klbf  = (__bf16*)(ws + OFF_BF) + BW_KL;

  const long long sQKVb = (long long)NPAD_ * K3_;
  const long long sHD   = (long long)MM_ * DH_;   // per-head ql/kl/a3v/zav stride
  const long long sSQ   = (long long)MM_ * MM_;   // per-head 256x256 stride

  // 1. LN + front pad
  ln_pad_kernel<<<B_ * NPAD_, 256, 0, st>>>(stream, lnw, lnb, pad);

  // 2. qkv = pad @ qkv_w^T (bf16 weights)
  gemm_awt_kernel<<<gemm_grid(NPAD_, K3_, B_), GEMM_BLOCK, 0, st>>>(
      pad, (long long)NPAD_ * C_, 0, C_,
      qkvw_bf, 0, 0, C_, nullptr,
      qkv, sQKVb, 0, K3_,
      NPAD_, K3_, C_, 1, 1.f, 0, 0, 0);

  // 3. landmarks (+ bf16 copy of kl)
  landmarks_kernel<<<eb(16LL * MM_ * DH_), 256, 0, st>>>(qkv, ql, kl, klbf);

  // 4. a2 = softmax(ql @ kl^T)
  gemm_awt_kernel<<<gemm_grid(MM_, MM_, 16), GEMM_BLOCK, 0, st>>>(
      ql, 8 * sHD, sHD, DH_,
      klbf, 8 * sHD, sHD, DH_, nullptr,
      a2, 8 * sSQ, sSQ, MM_,
      MM_, MM_, DH_, NH_, 1.f, 0, 0, 0);
  row_softmax_kernel<<<16 * MM_, 256, 0, st>>>(a2, MM_);

  // 5. z0 = a2^T / (max|rowsum| * max|colsum|)
  reset_scal_kernel<<<1, 256, 0, st>>>(scal);
  absrowmax_kernel<<<eb(16 * MM_), 256, 0, st>>>(a2, scal);
  abscolmax_kernel<<<eb(16 * MM_), 256, 0, st>>>(a2, scal);
  init_z_kernel<<<eb(16LL * MM_ * MM_), 256, 0, st>>>(a2, scal, zA);

  // 6. Newton iterations: z = 0.25 z (13I - t1(15I - t1(7I - t1))), t1 = a2 z
  float* zcur = zA;
  float* znxt = zB;
  const dim3 g256 = gemm_grid(MM_, MM_, 16);
  for (int it = 0; it < 6; ++it) {
    gemm_ab_kernel<<<g256, GEMM_BLOCK, 0, st>>>(            // t1 = a2 @ z
        a2,   8 * sSQ, sSQ, MM_, zcur, 8 * sSQ, sSQ, MM_,
        t1,   8 * sSQ, sSQ, MM_, MM_, MM_, MM_, NH_, 1.f, 0.f, 1.f, 0.f);
    gemm_ab_kernel<<<g256, GEMM_BLOCK, 0, st>>>(            // m2 = 15I - t1 @ (7I - t1)
        t1,   8 * sSQ, sSQ, MM_, t1,   8 * sSQ, sSQ, MM_,
        m2,   8 * sSQ, sSQ, MM_, MM_, MM_, MM_, NH_, -1.f, 7.f, -1.f, 15.f);
    gemm_ab_kernel<<<g256, GEMM_BLOCK, 0, st>>>(            // m3 = 13I - t1 @ m2
        t1,   8 * sSQ, sSQ, MM_, m2,   8 * sSQ, sSQ, MM_,
        m3,   8 * sSQ, sSQ, MM_, MM_, MM_, MM_, NH_, 1.f, 0.f, -1.f, 13.f);
    gemm_ab_kernel<<<g256, GEMM_BLOCK, 0, st>>>(            // z' = 0.25 z @ m3
        zcur, 8 * sSQ, sSQ, MM_, m3,   8 * sSQ, sSQ, MM_,
        znxt, 8 * sSQ, sSQ, MM_, MM_, MM_, MM_, NH_, 1.f, 0.f, 0.25f, 0.f);
    float* tmp = zcur; zcur = znxt; znxt = tmp;
  }

  // 7. a3 = softmax(ql @ k^T)  (256 x 8448), then a3v = a3 @ v
  gemm_awt_kernel<<<gemm_grid(MM_, NPAD_, 16), GEMM_BLOCK, 0, st>>>(
      ql, 8 * sHD, sHD, DH_,
      qkv + C_, sQKVb, (long long)DH_, K3_, nullptr,      // W rows = k[t], ldw = 1536
      big, 8LL * MM_ * NPAD_, (long long)MM_ * NPAD_, NPAD_,
      MM_, NPAD_, DH_, NH_, 1.f, 0, 0, 0);
  row_softmax_kernel<<<16 * MM_, 256, 0, st>>>(big, NPAD_);
  gemm_ab_kernel<<<gemm_grid(MM_, DH_, 16), GEMM_BLOCK, 0, st>>>(
      big, 8LL * MM_ * NPAD_, (long long)MM_ * NPAD_, NPAD_,
      qkv + 2 * C_, sQKVb, (long long)DH_, K3_,           // B rows = v[t], ldb = 1536
      a3v, 8 * sHD, sHD, DH_,
      MM_, DH_, NPAD_, NH_, 1.f, 0.f, 1.f, 0.f);

  // 8. zav = z @ a3v  (256x64)
  gemm_ab_kernel<<<gemm_grid(MM_, DH_, 16), GEMM_BLOCK, 0, st>>>(
      zcur, 8 * sSQ, sSQ, MM_,
      a3v, 8 * sHD, sHD, DH_,
      zav, 8 * sHD, sHD, DH_,
      MM_, DH_, MM_, NH_, 1.f, 0.f, 1.f, 0.f);

  // 9. a1 = softmax((q/8) @ kl^T)  (8448 x 256), then head-out = a1 @ zav  (into pad buffer)
  gemm_awt_kernel<<<gemm_grid(NPAD_, MM_, 16), GEMM_BLOCK, 0, st>>>(
      qkv, sQKVb, (long long)DH_, K3_,                    // A rows = q[t], lda = 1536
      klbf, 8 * sHD, sHD, DH_, nullptr,
      big, 8LL * NPAD_ * MM_, (long long)NPAD_ * MM_, MM_,
      NPAD_, MM_, DH_, NH_, 0.125f, 0, 0, 0);
  row_softmax_kernel<<<16 * NPAD_, 256, 0, st>>>(big, MM_);
  gemm_ab_kernel<<<gemm_grid(NPAD_, DH_, 16), GEMM_BLOCK, 0, st>>>(
      big, 8LL * NPAD_ * MM_, (long long)NPAD_ * MM_, MM_,
      zav, 8 * sHD, sHD, DH_,
      pad, (long long)NPAD_ * C_, (long long)DH_, C_,     // head h -> columns h*64..
      NPAD_, DH_, MM_, NH_, 1.f, 0.f, 1.f, 0.f);

  // 10. residual depthwise conv on v added to head-out
  resconv_kernel<<<eb((long long)B_ * NPAD_ * C_), 256, 0, st>>>(qkv, resw, pad);

  // 11. out-proj (bf16 weights), drop first 166 rows, accumulate into residual stream
  gemm_awt_kernel<<<gemm_grid(NPAD_, C_, B_), GEMM_BLOCK, 0, st>>>(
      pad, (long long)NPAD_ * C_, 0, C_,
      outw_bf, 0, 0, C_, outb,
      stream, (long long)N2_ * C_, 0, C_,
      NPAD_, C_, C_, 1, 1.f, 0, PADF_, 1);
}

extern "C" void kernel_launch(void* const* d_in, const int* in_sizes, int n_in,
                              void* d_out, int out_size, void* d_ws, size_t ws_size,
                              hipStream_t stream)
{
  const float* h      = (const float*)d_in[0];
  const float* fc1_w  = (const float*)d_in[1];
  const float* fc1_b  = (const float*)d_in[2];
  const float* cls    = (const float*)d_in[3];
  const float* ln1_w  = (const float*)d_in[4];
  const float* ln1_b  = (const float*)d_in[5];
  const float* qkv1_w = (const float*)d_in[6];
  const float* out1_w = (const float*)d_in[7];
  const float* out1_b = (const float*)d_in[8];
  const float* res1_w = (const float*)d_in[9];
  const float* ln2_w  = (const float*)d_in[10];
  const float* ln2_b  = (const float*)d_in[11];
  const float* qkv2_w = (const float*)d_in[12];
  const float* out2_w = (const float*)d_in[13];
  const float* out2_b = (const float*)d_in[14];
  const float* res2_w = (const float*)d_in[15];
  const float* p7_w   = (const float*)d_in[16];
  const float* p7_b   = (const float*)d_in[17];
  const float* p5_w   = (const float*)d_in[18];
  const float* p5_b   = (const float*)d_in[19];
  const float* p3_w   = (const float*)d_in[20];
  const float* p3_b   = (const float*)d_in[21];
  const float* lnf_w  = (const float*)d_in[22];
  const float* lnf_b  = (const float*)d_in[23];
  const float* fc2_w  = (const float*)d_in[24];
  const float* fc2_b  = (const float*)d_in[25];

  float* ws = (float*)d_ws;
  float* strm = ws + OFF_STREAM;
  float* big  = ws + OFF_BIG;
  __bf16* bfw = (__bf16*)(ws + OFF_BF);

  // convert stationary weights to bf16 once
  f32_to_bf16_kernel<<<eb(1048576), 256, 0, stream>>>(fc1_w,  bfw + BW_FC1,  1048576);
  f32_to_bf16_kernel<<<eb(786432),  256, 0, stream>>>(qkv1_w, bfw + BW_QKV1, 786432);
  f32_to_bf16_kernel<<<eb(786432),  256, 0, stream>>>(qkv2_w, bfw + BW_QKV2, 786432);
  f32_to_bf16_kernel<<<eb(262144),  256, 0, stream>>>(out1_w, bfw + BW_OUT1, 262144);
  f32_to_bf16_kernel<<<eb(262144),  256, 0, stream>>>(out2_w, bfw + BW_OUT2, 262144);

  // fc1: x = relu(h @ fc1_w^T + b) written into stream rows 1..8192 per batch
  gemm_awt_kernel<<<gemm_grid(N0_, C_, B_), GEMM_BLOCK, 0, stream>>>(
      h, (long long)N0_ * 2048, 0, 2048,
      bfw + BW_FC1, 0, 0, 2048, fc1_b,
      strm + C_, (long long)N2_ * C_, 0, C_,
      N0_, C_, 2048, 1, 1.f, 1, 0, 0);

  // cls token + wraparound rows
  fixup_kernel<<<eb((long long)B_ * 90 * C_), 256, 0, stream>>>(cls, strm);

  // attention block 1
  attention_block(ws, ln1_w, ln1_b, bfw + BW_QKV1, bfw + BW_OUT1, out1_b, res1_w, stream);

  // depthwise conv mixer on feat rows (7x7 + id + 5x5 + 3x3), then copy back
  dwconv_kernel<<<eb((long long)B_ * NSQ_ * C_), 256, 0, stream>>>(
      strm, p7_w, p7_b, p5_w, p5_b, p3_w, p3_b, big);
  copyback_kernel<<<eb((long long)B_ * NSQ_ * C_), 256, 0, stream>>>(big, strm);

  // attention block 2
  attention_block(ws, ln2_w, ln2_b, bfw + BW_QKV2, bfw + BW_OUT2, out2_b, res2_w, stream);

  // final LN(row 0) + fc2
  final_kernel<<<B_, 256, 0, stream>>>(strm, lnf_w, lnf_b, fc2_w, fc2_b, (float*)d_out);
}
